// AcidSAModule_58600533786798
// MI455X (gfx1250) — matrix operand order repaired and checked by reading, not compiled, not run on hardware
//
#include <hip/hip_runtime.h>
#include <hip/hip_bf16.h>

typedef __attribute__((ext_vector_type(16))) _Float16 v16h;
typedef __attribute__((ext_vector_type(8)))  _Float16 v8h;
typedef __attribute__((ext_vector_type(8)))  float    v8f;

static constexpr int N     = 32768;
static constexpr int M     = 8192;
static constexpr int D_X   = 128;
static constexpr int D_IN  = 131;   // D_X + 3
static constexpr int KP1   = 160;   // D_IN padded to multiple of 32
static constexpr int H     = 256;
static constexpr int D_OUT = 512;

// ---------------------------------------------------------------------------
// 1) Farthest point sampling: one persistent workgroup, 8191 dependent steps.
//    dists[] lives in global ws (L2-resident, 128KB); argmax via LDS tree.
// ---------------------------------------------------------------------------
__global__ __launch_bounds__(1024) void fps_kernel(const float* __restrict__ pos,
                                                   float* __restrict__ dists,
                                                   int* __restrict__ idx)
{
    __shared__ float s_val[1024];
    __shared__ int   s_idx[1024];
    __shared__ float s_last[3];
    const int t = threadIdx.x;

    for (int p = t; p < N; p += 1024) dists[p] = 3.402823e38f;
    if (t == 0) {
        idx[0] = 0;
        s_last[0] = pos[0]; s_last[1] = pos[1]; s_last[2] = pos[2];
    }
    __syncthreads();

    for (int it = 1; it < M; ++it) {
        const float lx = s_last[0], ly = s_last[1], lz = s_last[2];
        float best = -1.0f; int bidx = N;
        for (int p = t; p < N; p += 1024) {
            float dx = pos[3*p+0] - lx;
            float dy = pos[3*p+1] - ly;
            float dz = pos[3*p+2] - lz;
            float d  = dx*dx + dy*dy + dz*dz;
            float dm = fminf(dists[p], d);
            dists[p] = dm;
            if (dm > best || (dm == best && p < bidx)) { best = dm; bidx = p; }
        }
        s_val[t] = best; s_idx[t] = bidx;
        __syncthreads();
        for (int off = 512; off > 0; off >>= 1) {
            if (t < off) {
                float ov = s_val[t+off]; int oi = s_idx[t+off];
                if (ov > s_val[t] || (ov == s_val[t] && oi < s_idx[t])) {
                    s_val[t] = ov; s_idx[t] = oi;
                }
            }
            __syncthreads();
        }
        if (t == 0) {
            int nx = s_idx[0];
            idx[it] = nx;
            s_last[0] = pos[3*nx+0]; s_last[1] = pos[3*nx+1]; s_last[2] = pos[3*nx+2];
        }
        __syncthreads();
    }
}

// ---------------------------------------------------------------------------
// 2) Gather pos_dst = pos[idx], batch_out = (float)batch[idx]
// ---------------------------------------------------------------------------
__global__ void gather_kernel(const float* __restrict__ pos,
                              const int* __restrict__ idx,
                              const int* __restrict__ batch,
                              float* __restrict__ pos_dst,
                              float* __restrict__ batch_out)
{
    int m = blockIdx.x * blockDim.x + threadIdx.x;
    if (m >= M) return;
    int j = idx[m];
    pos_dst[3*m+0] = pos[3*j+0];
    pos_dst[3*m+1] = pos[3*j+1];
    pos_dst[3*m+2] = pos[3*j+2];
    batch_out[m]   = (float)batch[j];
}

// ---------------------------------------------------------------------------
// 3) Build f16 feature matrix A0[N][KP1]: [x | pos - pos_dst[residue]] + pad0
// ---------------------------------------------------------------------------
__global__ void build_features_kernel(const float* __restrict__ x,
                                      const float* __restrict__ pos,
                                      const float* __restrict__ pos_dst,
                                      const int* __restrict__ residue,
                                      _Float16* __restrict__ A0)
{
    int i = blockIdx.x * blockDim.x + threadIdx.x;
    if (i >= N * KP1) return;
    int e = i / KP1, c = i % KP1;
    float v;
    if (c < D_X) {
        v = x[(size_t)e * D_X + c];
    } else if (c < D_IN) {
        int r = residue[e];
        int d = c - D_X;
        v = pos[3*e + d] - pos_dst[3*r + d];
    } else {
        v = 0.0f;
    }
    A0[i] = (_Float16)v;
}

// ---------------------------------------------------------------------------
// 4) Weight convert+transpose: W[Kin][Out] f32 -> WT[Out][Kpad] f16 (pad 0)
// ---------------------------------------------------------------------------
__global__ void convert_wt_kernel(const float* __restrict__ W,
                                  _Float16* __restrict__ WT,
                                  int Kin, int Kpad, int Out)
{
    int i = blockIdx.x * blockDim.x + threadIdx.x;
    if (i >= Out * Kpad) return;
    int o = i / Kpad, k = i % Kpad;
    WT[i] = (_Float16)((k < Kin) ? W[(size_t)k * Out + o] : 0.0f);
}

// ---------------------------------------------------------------------------
// WMMA fragment helpers (v_wmma_f32_16x16x32_f16; A and B both use contiguous k = 16*lhalf + s)
// ---------------------------------------------------------------------------
__device__ inline v16h load_a_frag(const _Float16* __restrict__ A, int row,
                                   int kpad, int kt, int lhalf)
{
    return *(const v16h*)(A + (size_t)row * kpad + kt * 32 + lhalf * 16);
}

__device__ inline v16h load_b_frag_lds(const _Float16* sB, int n,
                                       int kpad, int kt, int lhalf)
{
    // sB row n holds the K-contiguous weights of output column n (in LDS).
    return *(const v16h*)(sB + (size_t)n * kpad + kt * 32 + lhalf * 16);
}

// ---------------------------------------------------------------------------
// 5) GEMM + bias + ReLU -> f16 activations.
//    Whole transposed weight panel (256 x KPAD f16) staged once per block in
//    LDS (80-128KB of the WGP's 320KB).  8 waves = 4 row-tiles x 2 col-halves;
//    each wave owns 32 rows x 128 cols: 2 A-fragments x 8 B-fragments ->
//    16 acc tiles (128 VGPRs), 16 WMMA per K-step with each B fragment reused
//    by 2 WMMAs (ds:wmma = 1:1, global:wmma = 1:4).
// ---------------------------------------------------------------------------
template<int KT, bool RELU>
__global__ __launch_bounds__(256) void gemm_relu_kernel(
    const _Float16* __restrict__ A, const _Float16* __restrict__ WT,
    const float* __restrict__ bias, _Float16* __restrict__ out)
{
    constexpr int KPAD = KT * 32;
    extern __shared__ char smem_raw[];
    _Float16* sB = (_Float16*)smem_raw;

    // Cooperative copy: full B panel (256 cols x KPAD) global -> LDS, 16B chunks
    {
        const uint4* src = (const uint4*)WT;
        uint4* dst = (uint4*)sB;
        const int nchunk = 256 * KPAD / 8;   // 8 halves per 16B chunk
        for (int i = threadIdx.x; i < nchunk; i += 256) dst[i] = src[i];
    }
    __syncthreads();

    const int lane  = threadIdx.x & 31;
    const int wave  = threadIdx.x >> 5;
    const int rtile = wave & 3;
    const int chalf = wave >> 2;
    const int rows  = (blockIdx.x * 4 + rtile) * 32;
    const int cbase = chalf * 128;
    const int lsub  = lane & 15;
    const int lhalf = lane >> 4;

    v8f acc[2][8];
#pragma unroll
    for (int r = 0; r < 2; ++r)
#pragma unroll
        for (int i = 0; i < 8; ++i) acc[r][i] = {};

    const int arow0 = rows + lsub;
    const int arow1 = rows + 16 + lsub;
    for (int kt = 0; kt < KT; ++kt) {
        if (kt + 1 < KT) {  // prefetch next K-step of A (global_prefetch_b8)
            __builtin_prefetch(A + (size_t)arow0 * KPAD + (kt + 1) * 32, 0, 0);
            __builtin_prefetch(A + (size_t)arow1 * KPAD + (kt + 1) * 32, 0, 0);
        }
        v16h a0 = load_a_frag(A, arow0, KPAD, kt, lhalf);
        v16h a1 = load_a_frag(A, arow1, KPAD, kt, lhalf);
#pragma unroll
        for (int nt = 0; nt < 8; ++nt) {
            v16h b = load_b_frag_lds(sB, cbase + nt * 16 + lsub, KPAD, kt, lhalf);
            acc[0][nt] = __builtin_amdgcn_wmma_f32_16x16x32_f16(
                false, a0, false, b, (short)0, acc[0][nt], false, false);
            acc[1][nt] = __builtin_amdgcn_wmma_f32_16x16x32_f16(
                false, a1, false, b, (short)0, acc[1][nt], false, false);
        }
    }

#pragma unroll
    for (int rt = 0; rt < 2; ++rt) {
#pragma unroll
        for (int nt = 0; nt < 8; ++nt) {
            const int col = cbase + nt * 16 + lsub;
            const float bv = bias[col];
#pragma unroll
            for (int v = 0; v < 8; ++v) {
                int row = rows + rt * 16 + v + 8 * lhalf;
                float val = acc[rt][nt][v] + bv;
                if (RELU) val = fmaxf(val, 0.0f);
                out[(size_t)row * 256 + col] = (_Float16)val;
            }
        }
    }
}

// ---------------------------------------------------------------------------
// 6) GEMM3 + bias, fused segment_max scatter via order-preserving u32 encode
//    + global atomicMax.  blockIdx.y selects 256-col half of D_OUT=512; that
//    half's weight panel (128KB) is staged in LDS; waves split 4x2 as above.
// ---------------------------------------------------------------------------
__device__ inline unsigned enc_f32(float f)
{
    unsigned u = __float_as_uint(f);
    return (u & 0x80000000u) ? ~u : (u | 0x80000000u);
}

__global__ __launch_bounds__(256) void gemm3_scatter_kernel(
    const _Float16* __restrict__ A, const _Float16* __restrict__ WT,
    const float* __restrict__ bias, const int* __restrict__ residue,
    unsigned* __restrict__ out)
{
    constexpr int KT = 8, KPAD = 256;
    extern __shared__ char smem_raw[];
    _Float16* sB = (_Float16*)smem_raw;

    const int half = blockIdx.y * 256;

    {   // stage this block's 256-col half of W3T in LDS
        const uint4* src = (const uint4*)(WT + (size_t)half * KPAD);
        uint4* dst = (uint4*)sB;
        const int nchunk = 256 * KPAD / 8;
        for (int i = threadIdx.x; i < nchunk; i += 256) dst[i] = src[i];
    }
    __syncthreads();

    const int lane  = threadIdx.x & 31;
    const int wave  = threadIdx.x >> 5;
    const int rtile = wave & 3;
    const int chalf = wave >> 2;
    const int rows  = (blockIdx.x * 4 + rtile) * 32;
    const int cbase = chalf * 128;           // within the LDS panel
    const int lsub  = lane & 15;
    const int lhalf = lane >> 4;

    v8f acc[2][8];
#pragma unroll
    for (int r = 0; r < 2; ++r)
#pragma unroll
        for (int i = 0; i < 8; ++i) acc[r][i] = {};

    const int arow0 = rows + lsub;
    const int arow1 = rows + 16 + lsub;
    for (int kt = 0; kt < KT; ++kt) {
        if (kt + 1 < KT) {
            __builtin_prefetch(A + (size_t)arow0 * KPAD + (kt + 1) * 32, 0, 0);
            __builtin_prefetch(A + (size_t)arow1 * KPAD + (kt + 1) * 32, 0, 0);
        }
        v16h a0 = load_a_frag(A, arow0, KPAD, kt, lhalf);
        v16h a1 = load_a_frag(A, arow1, KPAD, kt, lhalf);
#pragma unroll
        for (int nt = 0; nt < 8; ++nt) {
            v16h b = load_b_frag_lds(sB, cbase + nt * 16 + lsub, KPAD, kt, lhalf);
            acc[0][nt] = __builtin_amdgcn_wmma_f32_16x16x32_f16(
                false, a0, false, b, (short)0, acc[0][nt], false, false);
            acc[1][nt] = __builtin_amdgcn_wmma_f32_16x16x32_f16(
                false, a1, false, b, (short)0, acc[1][nt], false, false);
        }
    }

    int rr[2][8];
#pragma unroll
    for (int rt = 0; rt < 2; ++rt)
#pragma unroll
        for (int v = 0; v < 8; ++v)
            rr[rt][v] = residue[rows + rt * 16 + v + 8 * lhalf];

#pragma unroll
    for (int rt = 0; rt < 2; ++rt) {
#pragma unroll
        for (int nt = 0; nt < 8; ++nt) {
            const int col = half + cbase + nt * 16 + lsub;
            const float bv = bias[col];
#pragma unroll
            for (int v = 0; v < 8; ++v) {
                unsigned e = enc_f32(acc[rt][nt][v] + bv);
                atomicMax(&out[(size_t)rr[rt][v] * D_OUT + col], e);
            }
        }
    }
}

// ---------------------------------------------------------------------------
// 7) Output init (encoded -inf) and final decode (-inf -> 0)
// ---------------------------------------------------------------------------
__global__ void init_out_kernel(unsigned* __restrict__ out)
{
    int i = blockIdx.x * blockDim.x + threadIdx.x;
    if (i < M * D_OUT) out[i] = 0x007FFFFFu;  // enc(-inf)
}

__global__ void decode_out_kernel(float* __restrict__ out)
{
    int i = blockIdx.x * blockDim.x + threadIdx.x;
    if (i >= M * D_OUT) return;
    unsigned u  = ((const unsigned*)out)[i];
    unsigned ob = (u & 0x80000000u) ? (u ^ 0x80000000u) : ~u;
    out[i] = (ob == 0xFF800000u) ? 0.0f : __uint_as_float(ob);
}

// ---------------------------------------------------------------------------
// Launch
// ---------------------------------------------------------------------------
extern "C" void kernel_launch(void* const* d_in, const int* in_sizes, int n_in,
                              void* d_out, int out_size, void* d_ws, size_t ws_size,
                              hipStream_t stream)
{
    const float* x       = (const float*)d_in[0];
    const float* pos     = (const float*)d_in[1];
    const int*   residue = (const int*)  d_in[2];
    const int*   batch   = (const int*)  d_in[3];
    const float* W1      = (const float*)d_in[4];
    const float* b1      = (const float*)d_in[5];
    const float* W2      = (const float*)d_in[6];
    const float* b2      = (const float*)d_in[7];
    const float* W3      = (const float*)d_in[8];
    const float* b3      = (const float*)d_in[9];

    float* out       = (float*)d_out;                 // [M, D_OUT]
    float* pos_dst   = out + (size_t)M * D_OUT;       // [M, 3]
    float* batch_out = pos_dst + (size_t)M * 3;       // [M]

    char* ws = (char*)d_ws;
    size_t off = 0;
    auto alloc = [&](size_t bytes) -> void* {
        void* p = ws + off;
        off = (off + bytes + 255) & ~(size_t)255;
        return p;
    };
    _Float16* A0   = (_Float16*)alloc((size_t)N * KP1 * 2);
    _Float16* act1 = (_Float16*)alloc((size_t)N * H * 2);
    _Float16* act2 = (_Float16*)alloc((size_t)N * H * 2);
    _Float16* W1T  = (_Float16*)alloc((size_t)H * KP1 * 2);
    _Float16* W2T  = (_Float16*)alloc((size_t)H * H * 2);
    _Float16* W3T  = (_Float16*)alloc((size_t)D_OUT * H * 2);
    int*      idx  = (int*)     alloc((size_t)M * 4);
    float*    dist = (float*)   alloc((size_t)N * 4);

    // 1) FPS (serial dependency chain, one persistent workgroup)
    fps_kernel<<<1, 1024, 0, stream>>>(pos, dist, idx);

    // 2) Gather centers + batch
    gather_kernel<<<(M + 255) / 256, 256, 0, stream>>>(pos, idx, batch,
                                                       pos_dst, batch_out);

    // 3) Features -> f16 (padded K)
    build_features_kernel<<<(N * KP1 + 255) / 256, 256, 0, stream>>>(
        x, pos, pos_dst, residue, A0);

    // 4) Weights -> f16 transposed [out][Kpad]
    convert_wt_kernel<<<(H * KP1 + 255) / 256, 256, 0, stream>>>(W1, W1T, D_IN, KP1, H);
    convert_wt_kernel<<<(H * H + 255) / 256, 256, 0, stream>>>(W2, W2T, H, H, H);
    convert_wt_kernel<<<(D_OUT * H + 255) / 256, 256, 0, stream>>>(W3, W3T, H, H, D_OUT);

    // 5) MLP via WMMA (fp16 in, fp32 accumulate); B panels staged in LDS
    gemm_relu_kernel<KP1 / 32, true>
        <<<N / 128, 256, 256 * KP1 * 2, stream>>>(A0, W1T, b1, act1);
    gemm_relu_kernel<H / 32, true>
        <<<N / 128, 256, 256 * H * 2, stream>>>(act1, W2T, b2, act2);

    // 6) GEMM3 fused with segment-max scatter (per-half B panel in LDS)
    init_out_kernel<<<(M * D_OUT + 255) / 256, 256, 0, stream>>>((unsigned*)out);
    gemm3_scatter_kernel<<<dim3(N / 128, 2), 256, 256 * H * 2, stream>>>(
        act2, W3T, b3, residue, (unsigned*)out);

    // 7) Decode ordered-u32 back to float, empty segments -> 0
    decode_out_kernel<<<(M * D_OUT + 255) / 256, 256, 0, stream>>>(out);
}
